// GeometricMessagePassing_29515015258221
// MI455X (gfx1250) — matrix operand order repaired and checked by reading, not compiled, not run on hardware
//
#include <hip/hip_runtime.h>
#include <hip/hip_bf16.h>
#include <math.h>

constexpr int HIDC  = 128;   // hidden
constexpr int EDIMC = 32;    // edge feature dim
constexpr int NHEAD = 8;     // heads
constexpr int HDIM  = 16;    // head dim
constexpr int FFND  = 512;   // ffn dim

typedef __attribute__((ext_vector_type(16))) __bf16 v16bf;
typedef __attribute__((ext_vector_type(8)))  float  v8f;

// ---------- helpers ----------
__device__ __forceinline__ unsigned short f2bfbits(float f) {
  unsigned u = __float_as_uint(f);
  unsigned r = u + 0x7FFFu + ((u >> 16) & 1u);   // round-to-nearest-even
  return (unsigned short)(r >> 16);
}
__device__ __forceinline__ __bf16 us2bf(unsigned short s) {
  return __builtin_bit_cast(__bf16, s);
}
// order-preserving float<->uint encoding for atomicMax-based segment max
__device__ __forceinline__ unsigned fenc(float f) {
  unsigned b = __float_as_uint(f);
  return (b & 0x80000000u) ? ~b : (b | 0x80000000u);
}
__device__ __forceinline__ float fdec(unsigned u) {
  return __uint_as_float((u & 0x80000000u) ? (u & 0x7FFFFFFFu) : ~u);
}

// A fragment: 16x32 bf16 tile, row-major with leading dim ld.
// lane L: M = L%16 ; K = (L<16?0:16) + j
__device__ __forceinline__ v16bf load_a_frag(const unsigned short* A, int ld, int lane) {
  const unsigned short* p = A + (lane & 15) * ld + ((lane & 16) ? 16 : 0);
  v16bf a;
#pragma unroll
  for (int j = 0; j < 8; ++j)  a[j] = us2bf(p[j]);
#pragma unroll
  for (int j = 8; j < 16; ++j) a[j] = us2bf(p[j]);
  return a;
}
// B fragment: 32x16 (KxN); weights stored transposed (N-major): Wt[n*ld + k].
// lane L: N = L%16 ; K = (L<16?0:16) + j
__device__ __forceinline__ v16bf load_b_frag(const unsigned short* Wt, int ld, int lane) {
  const unsigned short* p = Wt + (lane & 15) * ld + ((lane & 16) ? 16 : 0);
  v16bf b;
#pragma unroll
  for (int j = 0; j < 16; ++j) b[j] = us2bf(p[j]);
  return b;
}

// ---------- setup kernels ----------
__global__ void k_cvt_x(const float* __restrict__ x, unsigned short* __restrict__ xb, int n) {
  int i = blockIdx.x * blockDim.x + threadIdx.x;
  if (i < n) xb[i] = f2bfbits(x[i]);
}
// W: K x Nout row-major (in,out)  ->  Wt: Nout x K bf16 (out,in)
__global__ void k_transpose_bf(const float* __restrict__ W, unsigned short* __restrict__ Wt,
                               int K, int Nout) {
  int i = blockIdx.x * blockDim.x + threadIdx.x;
  if (i < K * Nout) {
    int k = i / Nout, n = i % Nout;
    Wt[n * K + k] = f2bfbits(W[i]);
  }
}
__global__ void k_init(float* __restrict__ agg, unsigned* __restrict__ mx,
                       float* __restrict__ den, int nAgg, int nNH) {
  int i = blockIdx.x * blockDim.x + threadIdx.x;
  if (i < nAgg) agg[i] = 0.0f;
  if (i < nNH) { mx[i] = 0x007FFFFFu; /* fenc(-inf) */ den[i] = 0.0f; }
}

// ---------- fused q/k/v projection (WMMA) ----------
__global__ __launch_bounds__(256) void k_qkv(
    const unsigned short* __restrict__ xb,
    const unsigned short* __restrict__ qWt, const unsigned short* __restrict__ kWt,
    const unsigned short* __restrict__ vWt,
    const float* __restrict__ qb, const float* __restrict__ kbias,
    const float* __restrict__ vb,
    float* __restrict__ qf, float* __restrict__ kf, float* __restrict__ vf) {
  __shared__ unsigned short sA[16 * HIDC];
  const int tid = threadIdx.x, lane = tid & 31, wave = tid >> 5;
  const int r0 = blockIdx.x * 16;
  {
    int row = tid >> 4, cg = tid & 15;
    *(uint4*)&sA[row * HIDC + cg * 8] = *(const uint4*)&xb[(size_t)(r0 + row) * HIDC + cg * 8];
  }
  __syncthreads();
  const int n0 = wave * 16;
  v8f aq = {}, ak = {}, av = {};
#pragma unroll
  for (int kt = 0; kt < 4; ++kt) {
    const int kb0 = kt * 32;
    v16bf a  = load_a_frag(sA + kb0, HIDC, lane);
    v16bf b0 = load_b_frag(qWt + n0 * HIDC + kb0, HIDC, lane);
    aq = __builtin_amdgcn_wmma_f32_16x16x32_bf16(false, a, false, b0, (short)0, aq, false, false);
    v16bf b1 = load_b_frag(kWt + n0 * HIDC + kb0, HIDC, lane);
    ak = __builtin_amdgcn_wmma_f32_16x16x32_bf16(false, a, false, b1, (short)0, ak, false, false);
    v16bf b2 = load_b_frag(vWt + n0 * HIDC + kb0, HIDC, lane);
    av = __builtin_amdgcn_wmma_f32_16x16x32_bf16(false, a, false, b2, (short)0, av, false, false);
  }
  const int col = n0 + (lane & 15);
  const int rb = r0 + ((lane & 16) ? 8 : 0);
#pragma unroll
  for (int i = 0; i < 8; ++i) {
    const size_t idx = (size_t)(rb + i) * HIDC + col;
    qf[idx] = aq[i] + qb[col];
    kf[idx] = ak[i] + kbias[col];
    vf[idx] = av[i] + vb[col];
  }
}

// ---------- edge bias: edge_attr @ eW + eb ----------
__global__ __launch_bounds__(256) void k_edge_bias(
    const float* __restrict__ ea, const float* __restrict__ eW,
    const float* __restrict__ eb, float* __restrict__ scores, int E) {
  __shared__ float sW[EDIMC * NHEAD];
  __shared__ float sB[NHEAD];
  int tid = threadIdx.x;
  if (tid < EDIMC * NHEAD) sW[tid] = eW[tid];
  if (tid < NHEAD) sB[tid] = eb[tid];
  __syncthreads();
  int e = blockIdx.x * blockDim.x + tid;
  if (e >= E) return;
  float f[EDIMC];
#pragma unroll
  for (int d = 0; d < EDIMC; ++d) f[d] = ea[(size_t)e * EDIMC + d];
#pragma unroll
  for (int h = 0; h < NHEAD; ++h) {
    float s = sB[h];
#pragma unroll
    for (int d = 0; d < EDIMC; ++d) s += f[d] * sW[d * NHEAD + h];
    scores[(size_t)e * NHEAD + h] = s;
  }
}

// ---------- per-edge scores + segment max ----------
__global__ __launch_bounds__(256) void k_scores(
    const float* __restrict__ qf, const float* __restrict__ kf,
    const int* __restrict__ ei, float* __restrict__ scores,
    unsigned* __restrict__ mx, int E) {
  int g = blockIdx.x * blockDim.x + threadIdx.x;
  if (g >= E * NHEAD) return;
  int e = g >> 3, h = g & 7;
  int src = ei[e], dst = ei[E + e];
  const float4* qp = (const float4*)(qf + (size_t)src * HIDC + h * HDIM);
  const float4* kp = (const float4*)(kf + (size_t)dst * HIDC + h * HDIM);
  float dot = 0.0f;
#pragma unroll
  for (int u = 0; u < 4; ++u) {
    float4 a = qp[u], c = kp[u];
    dot += a.x * c.x + a.y * c.y + a.z * c.z + a.w * c.w;
  }
  float s = dot * 0.25f + scores[g];   // 1/sqrt(16) + edge bias
  scores[g] = s;
  atomicMax(&mx[(size_t)dst * NHEAD + h], fenc(s));
}

// ---------- exp + segment sum ----------
__global__ __launch_bounds__(256) void k_expden(
    const int* __restrict__ ei, float* __restrict__ scores,
    const unsigned* __restrict__ mx, float* __restrict__ den, int E) {
  int g = blockIdx.x * blockDim.x + threadIdx.x;
  if (g >= E * NHEAD) return;
  int e = g >> 3, h = g & 7;
  int dst = ei[E + e];
  float m = fdec(mx[(size_t)dst * NHEAD + h]);
  if (isinf(m)) m = 0.0f;
  float ex = __expf(scores[g] - m);
  scores[g] = ex;
  atomicAdd(&den[(size_t)dst * NHEAD + h], ex);
}

// ---------- weighted aggregation (scatter) ----------
__global__ __launch_bounds__(256) void k_agg(
    const int* __restrict__ ei, const float* __restrict__ scores,
    const float* __restrict__ den, const float* __restrict__ vf,
    float* __restrict__ agg, int E) {
  int g = blockIdx.x * blockDim.x + threadIdx.x;
  if (g >= E * NHEAD) return;
  int e = g >> 3, h = g & 7;
  int src = ei[e], dst = ei[E + e];
  float attn = scores[g] / fmaxf(den[(size_t)dst * NHEAD + h], 1e-8f);
  const float* vp = vf + (size_t)src * HIDC + h * HDIM;
  float* ap = agg + (size_t)dst * HIDC + h * HDIM;
#pragma unroll
  for (int d = 0; d < HDIM; ++d) atomicAdd(&ap[d], vp[d] * attn);
}

// ---------- out-proj + residual + LN1 (WMMA) ----------
__global__ __launch_bounds__(256) void k_attnout_ln1(
    const float* __restrict__ agg, const float* __restrict__ x,
    const unsigned short* __restrict__ oWt, const float* __restrict__ ob,
    const float* __restrict__ g, const float* __restrict__ b,
    float* __restrict__ x1f, unsigned short* __restrict__ x1b) {
  __shared__ unsigned short sA[16 * HIDC];
  __shared__ float sO[16 * HIDC];
  __shared__ float redS[256], redQ[256];
  __shared__ float muA[16], rsA[16];
  const int tid = threadIdx.x, lane = tid & 31, wave = tid >> 5;
  const int r0 = blockIdx.x * 16;
  {
    int row = tid >> 4, cg = tid & 15;
#pragma unroll
    for (int j = 0; j < 8; ++j)
      sA[row * HIDC + cg * 8 + j] = f2bfbits(agg[(size_t)(r0 + row) * HIDC + cg * 8 + j]);
  }
  __syncthreads();
  const int n0 = wave * 16;
  v8f acc = {};
#pragma unroll
  for (int kt = 0; kt < 4; ++kt) {
    const int kb0 = kt * 32;
    v16bf a  = load_a_frag(sA + kb0, HIDC, lane);
    v16bf bb = load_b_frag(oWt + n0 * HIDC + kb0, HIDC, lane);
    acc = __builtin_amdgcn_wmma_f32_16x16x32_bf16(false, a, false, bb, (short)0, acc, false, false);
  }
  const int col = n0 + (lane & 15);
  const int rloc = (lane & 16) ? 8 : 0;
#pragma unroll
  for (int i = 0; i < 8; ++i) {
    int r = rloc + i;
    sO[r * HIDC + col] = acc[i] + ob[col] + x[(size_t)(r0 + r) * HIDC + col];
  }
  __syncthreads();
  const int row = tid >> 4, t = tid & 15;
  float s = 0.0f, q = 0.0f;
#pragma unroll
  for (int j = 0; j < 8; ++j) {
    float v = sO[row * HIDC + t * 8 + j];
    s += v; q += v * v;
  }
  redS[tid] = s; redQ[tid] = q;
  __syncthreads();
  if (t == 0) {
    float S = 0.0f, Q = 0.0f;
#pragma unroll
    for (int i = 0; i < 16; ++i) { S += redS[row * 16 + i]; Q += redQ[row * 16 + i]; }
    float mu = S * (1.0f / HIDC);
    float var = Q * (1.0f / HIDC) - mu * mu;
    muA[row] = mu; rsA[row] = rsqrtf(var + 1e-5f);
  }
  __syncthreads();
#pragma unroll
  for (int j = 0; j < 8; ++j) {
    int c = t * 8 + j;
    float y = (sO[row * HIDC + c] - muA[row]) * rsA[row] * g[c] + b[c];
    size_t idx = (size_t)(r0 + row) * HIDC + c;
    x1f[idx] = y;
    x1b[idx] = f2bfbits(y);
  }
}

// ---------- FFN1 + LN + exact GELU (WMMA) ----------
__global__ __launch_bounds__(256) void k_ffn1(
    const unsigned short* __restrict__ x1b,
    const unsigned short* __restrict__ f1Wt, const float* __restrict__ f1b,
    const float* __restrict__ g, const float* __restrict__ b,
    unsigned short* __restrict__ hb) {
  __shared__ unsigned short sA[16 * HIDC];
  __shared__ float sO[16 * FFND];
  __shared__ float redS[256], redQ[256];
  __shared__ float muA[16], rsA[16];
  const int tid = threadIdx.x, lane = tid & 31, wave = tid >> 5;
  const int r0 = blockIdx.x * 16;
  {
    int row = tid >> 4, cg = tid & 15;
    *(uint4*)&sA[row * HIDC + cg * 8] = *(const uint4*)&x1b[(size_t)(r0 + row) * HIDC + cg * 8];
  }
  __syncthreads();
  const int rloc = (lane & 16) ? 8 : 0;
#pragma unroll
  for (int ct = 0; ct < 4; ++ct) {
    const int n0 = wave * 64 + ct * 16;
    v8f acc = {};
#pragma unroll
    for (int kt = 0; kt < 4; ++kt) {
      const int kb0 = kt * 32;
      v16bf a  = load_a_frag(sA + kb0, HIDC, lane);
      v16bf bb = load_b_frag(f1Wt + n0 * HIDC + kb0, HIDC, lane);
      acc = __builtin_amdgcn_wmma_f32_16x16x32_bf16(false, a, false, bb, (short)0, acc, false, false);
    }
    const int col = n0 + (lane & 15);
#pragma unroll
    for (int i = 0; i < 8; ++i)
      sO[(rloc + i) * FFND + col] = acc[i] + f1b[col];
  }
  __syncthreads();
  const int row = tid >> 4, t = tid & 15;
  float s = 0.0f, q = 0.0f;
#pragma unroll
  for (int j = 0; j < 32; ++j) {
    float v = sO[row * FFND + t * 32 + j];
    s += v; q += v * v;
  }
  redS[tid] = s; redQ[tid] = q;
  __syncthreads();
  if (t == 0) {
    float S = 0.0f, Q = 0.0f;
#pragma unroll
    for (int i = 0; i < 16; ++i) { S += redS[row * 16 + i]; Q += redQ[row * 16 + i]; }
    float mu = S * (1.0f / FFND);
    float var = Q * (1.0f / FFND) - mu * mu;
    muA[row] = mu; rsA[row] = rsqrtf(var + 1e-5f);
  }
  __syncthreads();
#pragma unroll
  for (int j = 0; j < 32; ++j) {
    int c = t * 32 + j;
    float y = (sO[row * FFND + c] - muA[row]) * rsA[row] * g[c] + b[c];
    y = 0.5f * y * (1.0f + erff(y * 0.70710678118f));   // exact GELU
    hb[(size_t)(r0 + row) * FFND + c] = f2bfbits(y);
  }
}

// ---------- FFN2 + residual + LN2 -> output (WMMA) ----------
__global__ __launch_bounds__(256) void k_ffn2(
    const unsigned short* __restrict__ hb,
    const unsigned short* __restrict__ f2Wt, const float* __restrict__ f2b,
    const float* __restrict__ x1f,
    const float* __restrict__ g, const float* __restrict__ b,
    float* __restrict__ out) {
  __shared__ unsigned short sA[16 * FFND];
  __shared__ float sO[16 * HIDC];
  __shared__ float redS[256], redQ[256];
  __shared__ float muA[16], rsA[16];
  const int tid = threadIdx.x, lane = tid & 31, wave = tid >> 5;
  const int r0 = blockIdx.x * 16;
  {
    int row = tid >> 4, cg = tid & 15;
#pragma unroll
    for (int u = 0; u < 4; ++u)
      *(uint4*)&sA[row * FFND + cg * 32 + u * 8] =
          *(const uint4*)&hb[(size_t)(r0 + row) * FFND + cg * 32 + u * 8];
  }
  __syncthreads();
  const int n0 = wave * 16;
  v8f acc = {};
#pragma unroll
  for (int kt = 0; kt < 16; ++kt) {
    const int kb0 = kt * 32;
    v16bf a  = load_a_frag(sA + kb0, FFND, lane);
    v16bf bb = load_b_frag(f2Wt + n0 * FFND + kb0, FFND, lane);
    acc = __builtin_amdgcn_wmma_f32_16x16x32_bf16(false, a, false, bb, (short)0, acc, false, false);
  }
  const int col = n0 + (lane & 15);
  const int rloc = (lane & 16) ? 8 : 0;
#pragma unroll
  for (int i = 0; i < 8; ++i) {
    int r = rloc + i;
    sO[r * HIDC + col] = acc[i] + f2b[col] + x1f[(size_t)(r0 + r) * HIDC + col];
  }
  __syncthreads();
  const int row = tid >> 4, t = tid & 15;
  float s = 0.0f, q = 0.0f;
#pragma unroll
  for (int j = 0; j < 8; ++j) {
    float v = sO[row * HIDC + t * 8 + j];
    s += v; q += v * v;
  }
  redS[tid] = s; redQ[tid] = q;
  __syncthreads();
  if (t == 0) {
    float S = 0.0f, Q = 0.0f;
#pragma unroll
    for (int i = 0; i < 16; ++i) { S += redS[row * 16 + i]; Q += redQ[row * 16 + i]; }
    float mu = S * (1.0f / HIDC);
    float var = Q * (1.0f / HIDC) - mu * mu;
    muA[row] = mu; rsA[row] = rsqrtf(var + 1e-5f);
  }
  __syncthreads();
#pragma unroll
  for (int j = 0; j < 8; ++j) {
    int c = t * 8 + j;
    float y = (sO[row * HIDC + c] - muA[row]) * rsA[row] * g[c] + b[c];
    out[(size_t)(r0 + row) * HIDC + c] = y;
  }
}

// ---------- host launcher ----------
extern "C" void kernel_launch(void* const* d_in, const int* in_sizes, int n_in,
                              void* d_out, int out_size, void* d_ws, size_t ws_size,
                              hipStream_t stream) {
  const float* x    = (const float*)d_in[0];
  const int*   ei   = (const int*)d_in[1];
  const float* ea   = (const float*)d_in[2];
  const float* qW   = (const float*)d_in[3];
  const float* qb   = (const float*)d_in[4];
  const float* kW   = (const float*)d_in[5];
  const float* kb   = (const float*)d_in[6];
  const float* vW   = (const float*)d_in[7];
  const float* vb   = (const float*)d_in[8];
  const float* eW   = (const float*)d_in[9];
  const float* eb   = (const float*)d_in[10];
  const float* oW   = (const float*)d_in[11];
  const float* ob   = (const float*)d_in[12];
  const float* ln1g = (const float*)d_in[13];
  const float* ln1b = (const float*)d_in[14];
  const float* f1W  = (const float*)d_in[15];
  const float* f1b  = (const float*)d_in[16];
  const float* lnfg = (const float*)d_in[17];
  const float* lnfb = (const float*)d_in[18];
  const float* f2W  = (const float*)d_in[19];
  const float* f2b  = (const float*)d_in[20];
  const float* ln2g = (const float*)d_in[21];
  const float* ln2b = (const float*)d_in[22];
  float* out = (float*)d_out;

  const int N = in_sizes[0] / HIDC;
  const int E = in_sizes[1] / 2;

  // workspace carve-out (256B aligned slices)
  char* w = (char*)d_ws;
  size_t off = 0;
  auto take = [&](size_t bytes) -> void* {
    void* p = w + off;
    off = (off + bytes + 255) & ~(size_t)255;
    return p;
  };
  unsigned short* xb   = (unsigned short*)take((size_t)N * HIDC * 2);
  unsigned short* qWt  = (unsigned short*)take((size_t)HIDC * HIDC * 2);
  unsigned short* kWt  = (unsigned short*)take((size_t)HIDC * HIDC * 2);
  unsigned short* vWt  = (unsigned short*)take((size_t)HIDC * HIDC * 2);
  unsigned short* oWt  = (unsigned short*)take((size_t)HIDC * HIDC * 2);
  unsigned short* f1Wt = (unsigned short*)take((size_t)HIDC * FFND * 2);
  unsigned short* f2Wt = (unsigned short*)take((size_t)FFND * HIDC * 2);
  float* qf     = (float*)take((size_t)N * HIDC * 4);
  float* kf     = (float*)take((size_t)N * HIDC * 4);
  float* vf     = (float*)take((size_t)N * HIDC * 4);
  float* scores = (float*)take((size_t)E * NHEAD * 4);
  unsigned* mx  = (unsigned*)take((size_t)N * NHEAD * 4);
  float* den    = (float*)take((size_t)N * NHEAD * 4);
  float* agg    = (float*)take((size_t)N * HIDC * 4);
  float* x1f    = (float*)take((size_t)N * HIDC * 4);
  unsigned short* x1b = (unsigned short*)take((size_t)N * HIDC * 2);
  unsigned short* hb  = (unsigned short*)take((size_t)N * FFND * 2);

  const int T = 256;
  // setup: bf16 conversions + weight transposes
  k_cvt_x<<<(N * HIDC + T - 1) / T, T, 0, stream>>>(x, xb, N * HIDC);
  k_transpose_bf<<<(HIDC * HIDC + T - 1) / T, T, 0, stream>>>(qW, qWt, HIDC, HIDC);
  k_transpose_bf<<<(HIDC * HIDC + T - 1) / T, T, 0, stream>>>(kW, kWt, HIDC, HIDC);
  k_transpose_bf<<<(HIDC * HIDC + T - 1) / T, T, 0, stream>>>(vW, vWt, HIDC, HIDC);
  k_transpose_bf<<<(HIDC * HIDC + T - 1) / T, T, 0, stream>>>(oW, oWt, HIDC, HIDC);
  k_transpose_bf<<<(HIDC * FFND + T - 1) / T, T, 0, stream>>>(f1W, f1Wt, HIDC, FFND);
  k_transpose_bf<<<(FFND * HIDC + T - 1) / T, T, 0, stream>>>(f2W, f2Wt, FFND, HIDC);
  k_init<<<(N * HIDC + T - 1) / T, T, 0, stream>>>(agg, mx, den, N * HIDC, N * NHEAD);

  const int nblk = N / 16;   // N = 50000 -> 3125 row tiles
  k_qkv<<<nblk, T, 0, stream>>>(xb, qWt, kWt, vWt, qb, kb, vb, qf, kf, vf);

  k_edge_bias<<<(E + T - 1) / T, T, 0, stream>>>(ea, eW, eb, scores, E);
  const int EH = E * NHEAD;
  k_scores<<<(EH + T - 1) / T, T, 0, stream>>>(qf, kf, ei, scores, mx, E);
  k_expden<<<(EH + T - 1) / T, T, 0, stream>>>(ei, scores, mx, den, E);
  k_agg<<<(EH + T - 1) / T, T, 0, stream>>>(ei, scores, den, vf, agg, E);

  k_attnout_ln1<<<nblk, T, 0, stream>>>(agg, x, oWt, ob, ln1g, ln1b, x1f, x1b);
  k_ffn1<<<nblk, T, 0, stream>>>(x1b, f1Wt, f1b, lnfg, lnfb, hb);
  k_ffn2<<<nblk, T, 0, stream>>>(hb, f2Wt, f2b, x1f, ln2g, ln2b, out);
}
